// VanillaGNN_8160437862403
// MI455X (gfx1250) — matrix operand order repaired and checked by reading, not compiled, not run on hardware
//
#include <hip/hip_runtime.h>
#include <hip/hip_bf16.h>

typedef __attribute__((ext_vector_type(16))) _Float16 v16h;
typedef __attribute__((ext_vector_type(8)))  _Float16 v8h;
typedef __attribute__((ext_vector_type(8)))  float    v8f;

#define N_NODES 50000
#define IN_DIM  130
#define HID     128
#define KPAD1   160   // IN_DIM padded up to multiple of 32

// ---------------- degree / norm ----------------
__global__ void k_deg_init(float* deg) {
    int i = blockIdx.x * blockDim.x + threadIdx.x;
    if (i < N_NODES) deg[i] = 1.0f;          // self-loop contributes 1
}

__global__ void k_deg_accum(const int* __restrict__ dstIdx, float* __restrict__ deg, int E) {
    int e = blockIdx.x * blockDim.x + threadIdx.x;
    if (e < E) atomicAdd(&deg[dstIdx[e]], 1.0f);
}

__global__ void k_rsqrt(float* d) {
    int i = blockIdx.x * blockDim.x + threadIdx.x;
    if (i < N_NODES) d[i] = rsqrtf(d[i]);    // deg >= 1 always
}

// ---------------- conversions ----------------
// x[N][IN_DIM] f32 -> Xh[N][KPAD1] f16 (zero padded)
__global__ void k_cvt_x(const float* __restrict__ x, _Float16* __restrict__ xh) {
    int idx = blockIdx.x * blockDim.x + threadIdx.x;
    if (idx >= N_NODES * KPAD1) return;
    int i = idx / KPAD1, k = idx - i * KPAD1;
    xh[idx] = (k < IN_DIM) ? (_Float16)x[(size_t)i * IN_DIM + k] : (_Float16)0.0f;
}

// W1[IN_DIM][HID] f32 -> W1t[HID][KPAD1] f16 (transposed, zero padded)
__global__ void k_cvt_w1(const float* __restrict__ W, _Float16* __restrict__ Wt) {
    int idx = blockIdx.x * blockDim.x + threadIdx.x;
    if (idx >= HID * KPAD1) return;
    int n = idx / KPAD1, k = idx - n * KPAD1;
    Wt[idx] = (k < IN_DIM) ? (_Float16)W[(size_t)k * HID + n] : (_Float16)0.0f;
}

// W2[HID][HID] f32 -> W2t[HID][HID] f16 (transposed)
__global__ void k_cvt_w2(const float* __restrict__ W, _Float16* __restrict__ Wt) {
    int idx = blockIdx.x * blockDim.x + threadIdx.x;
    if (idx >= HID * HID) return;
    int n = idx / HID, k = idx - n * HID;
    Wt[idx] = (_Float16)W[(size_t)k * HID + n];
}

// ---------------- WMMA GEMM: C[N][HID] = A[N][KPAD] * Bt[HID][KPAD]^T ----------------
// One wave computes a 16x16 tile; 8 waves per block cover a 16-row x 128-col stripe.
template <int KPAD>
__global__ void __launch_bounds__(256) k_gemm_wmma(const _Float16* __restrict__ A,
                                                   const _Float16* __restrict__ Bt,
                                                   float* __restrict__ C) {
    const int lane = threadIdx.x;          // 0..31
    const int nt   = threadIdx.y;          // 0..7  (column tile)
    const int mt   = blockIdx.x;           // row tile (N_NODES/16 = 3125 exact)
    const int rc   = lane & 15;
    const int kh8  = (lane >> 4) * 8;      // A-matrix K sub-offset (ISA 7.12.2)
    const int kh16 = (lane >> 4) * 16;     // unused (B also uses kh8)

    const _Float16* ap = A  + (size_t)(mt * 16 + rc) * KPAD + kh8;
    const _Float16* bp = Bt + (size_t)(nt * 16 + rc) * KPAD + kh8;

    v8f c = {};
#pragma unroll
    for (int kb = 0; kb < KPAD / 32; ++kb) {
        v16h a, b;
        // A 16x32 f16 layout: VGPR0-3 <- K[kh8..kh8+7], VGPR4-7 <- K[16+kh8..16+kh8+7]
        ((v8h*)&a)[0] = *(const v8h*)(ap + kb * 32);
        ((v8h*)&a)[1] = *(const v8h*)(ap + kb * 32 + 16);
        // B 32x16 f16 layout: lane holds col n, K[kh8..kh8+7], K[16+kh8..16+kh8+7]
        ((v8h*)&b)[0] = *(const v8h*)(bp + kb * 32);
        ((v8h*)&b)[1] = *(const v8h*)(bp + kb * 32 + 16);
        c = __builtin_amdgcn_wmma_f32_16x16x32_f16(
            /*neg_a=*/false, a, /*neg_b=*/false, b,
            /*c_mod=*/(short)0, c, /*reuse_a=*/false, /*reuse_b=*/false);
    }

    // D layout: VGPR r -> M = (lane>>4)*8 + r, N = lane&15
    float* cp = C + (size_t)(mt * 16 + kh8) * HID + nt * 16 + rc;
#pragma unroll
    for (int r = 0; r < 8; ++r) cp[(size_t)r * HID] = c[r];
}

// ---------------- aggregation ----------------
// out[i][f] = H[i][f] * dinv[i]^2 (self loop) + bias[f]
__global__ void k_agg_init(const float* __restrict__ H, const float* __restrict__ dinv,
                           const float* __restrict__ bias, float* __restrict__ out) {
    int idx = blockIdx.x * blockDim.x + threadIdx.x;
    if (idx >= N_NODES * HID) return;
    int i = idx >> 7, f = idx & 127;
    float di = dinv[i];
    out[idx] = H[idx] * di * di + bias[f];
}

// one wave32 per edge: gather 128 f32 (float4/lane), scale, 4 f32 atomics/lane
__global__ void __launch_bounds__(256) k_scatter(const int* __restrict__ srcIdx,
                                                 const int* __restrict__ dstIdx,
                                                 const float* __restrict__ dinv,
                                                 const float* __restrict__ H,
                                                 float* __restrict__ out, int E) {
    int e = blockIdx.x * 8 + (threadIdx.x >> 5);
    if (e >= E) return;
    int lane = threadIdx.x & 31;
    int s = srcIdx[e], d = dstIdx[e];
    float norm = dinv[s] * dinv[d];
    float4 v = ((const float4*)(H + (size_t)s * HID))[lane];
    float* op = out + (size_t)d * HID + lane * 4;
    atomicAdd(op + 0, v.x * norm);
    atomicAdd(op + 1, v.y * norm);
    atomicAdd(op + 2, v.z * norm);
    atomicAdd(op + 3, v.w * norm);
}

__global__ void k_relu_cvt(const float* __restrict__ in, _Float16* __restrict__ outh) {
    int idx = blockIdx.x * blockDim.x + threadIdx.x;
    if (idx >= N_NODES * HID) return;
    float v = in[idx];
    outh[idx] = (_Float16)(v > 0.0f ? v : 0.0f);
}

__global__ void k_relu_inplace(float* io) {
    int idx = blockIdx.x * blockDim.x + threadIdx.x;
    if (idx >= N_NODES * HID) return;
    float v = io[idx];
    io[idx] = v > 0.0f ? v : 0.0f;
}

// ---------------- launch ----------------
extern "C" void kernel_launch(void* const* d_in, const int* in_sizes, int n_in,
                              void* d_out, int out_size, void* d_ws, size_t ws_size,
                              hipStream_t stream) {
    const float* x  = (const float*)d_in[0];
    const int*   ei = (const int*)d_in[1];
    const float* W1 = (const float*)d_in[2];
    const float* b1 = (const float*)d_in[3];
    const float* W2 = (const float*)d_in[4];
    const float* b2 = (const float*)d_in[5];
    float* out = (float*)d_out;

    const int E = in_sizes[1] / 2;
    const int* srcIdx = ei;
    const int* dstIdx = ei + E;

    // carve workspace (256B aligned partitions)
    char* ws = (char*)d_ws;
    auto alloc = [&](size_t bytes) -> char* {
        char* p = ws;
        ws += (bytes + 255) & ~(size_t)255;
        return p;
    };
    float*    dinv = (float*)   alloc((size_t)N_NODES * 4);
    _Float16* Xh   = (_Float16*)alloc((size_t)N_NODES * KPAD1 * 2);
    _Float16* W1t  = (_Float16*)alloc((size_t)HID * KPAD1 * 2);
    _Float16* W2t  = (_Float16*)alloc((size_t)HID * HID * 2);
    float*    Hg   = (float*)   alloc((size_t)N_NODES * HID * 4);   // gemm output
    _Float16* H1h  = (_Float16*)alloc((size_t)N_NODES * HID * 2);   // relu(layer1) in f16

    const int T = 256;
    dim3 gemmBlk(32, 8);
    dim3 gemmGrd(N_NODES / 16);   // 3125, exact

    // degrees -> dinv
    k_deg_init <<<(N_NODES + T - 1) / T, T, 0, stream>>>(dinv);
    k_deg_accum<<<(E + T - 1) / T, T, 0, stream>>>(dstIdx, dinv, E);
    k_rsqrt    <<<(N_NODES + T - 1) / T, T, 0, stream>>>(dinv);

    // conversions
    k_cvt_x <<<(N_NODES * KPAD1 + T - 1) / T, T, 0, stream>>>(x, Xh);
    k_cvt_w1<<<(HID * KPAD1 + T - 1) / T, T, 0, stream>>>(W1, W1t);
    k_cvt_w2<<<(HID * HID + T - 1) / T, T, 0, stream>>>(W2, W2t);

    // ---- layer 1 ----
    k_gemm_wmma<KPAD1><<<gemmGrd, gemmBlk, 0, stream>>>(Xh, W1t, Hg);
    k_agg_init<<<(N_NODES * HID + T - 1) / T, T, 0, stream>>>(Hg, dinv, b1, out); // out as f32 scratch
    k_scatter <<<(E + 7) / 8, T, 0, stream>>>(srcIdx, dstIdx, dinv, Hg, out, E);
    k_relu_cvt<<<(N_NODES * HID + T - 1) / T, T, 0, stream>>>(out, H1h);

    // ---- layer 2 ----
    k_gemm_wmma<HID><<<gemmGrd, gemmBlk, 0, stream>>>(H1h, W2t, Hg);
    k_agg_init<<<(N_NODES * HID + T - 1) / T, T, 0, stream>>>(Hg, dinv, b2, out);
    k_scatter <<<(E + 7) / 8, T, 0, stream>>>(srcIdx, dstIdx, dinv, Hg, out, E);
    k_relu_inplace<<<(N_NODES * HID + T - 1) / T, T, 0, stream>>>(out);
}
